// FirstBlockVisualizer_60662118089291
// MI455X (gfx1250) — matrix operand order repaired and checked by reading, not compiled, not run on hardware
//
#include <hip/hip_runtime.h>
#include <hip/hip_bf16.h>
#include <math.h>

// ---------------------------------------------------------------------------
// Restormer first block for MI455X (gfx1250), wave32 + WMMA bf16 16x16x32.
// All 1x1 convs + channel attention run as v_wmma_f32_16x16x32_bf16 GEMMs.
// Weight panels are DMA'd into LDS once per workgroup via the Tensor Data
// Mover (tensor_load_to_lds + s_wait_tensorcnt), so the vector-memory path
// is dedicated to streaming activations (workload is bandwidth-bound:
// ~60 GFLOP vs ~1.5 GB of traffic at 23.3 TB/s).
// Layouts are chosen so every WMMA operand is a contiguous 16B/32B-per-lane
// load (no transposes):
//   - weights prepacked row-major [Mp][Kp] bf16 (K zero-padded to mult of 32)
//   - GEMM activations channel-last [n][Kp] bf16
//   - q,k planar [c][n] bf16 (K-dim = n is contiguous for A and B frags)
// po_w @ attn is folded into one 48x48 matrix so attn@v + project_out is a
// single GEMM sweep.  All reductions are fixed-order (deterministic).
// Workspace requirement: ~496 MB.
// ---------------------------------------------------------------------------

typedef __bf16 bf16_t;
typedef __attribute__((ext_vector_type(16))) __bf16 v16bf;
typedef __attribute__((ext_vector_type(8)))  float  v8f;
typedef unsigned int u32x4 __attribute__((ext_vector_type(4)));
typedef int          i32x4 __attribute__((ext_vector_type(4)));
typedef int          i32x8 __attribute__((ext_vector_type(8)));

#define HDIM 512
#define WDIM 512
#define NPIX (HDIM * WDIM)   // 262144
#define NB   2

#if __has_builtin(__builtin_amdgcn_tensor_load_to_lds) && \
    __has_builtin(__builtin_amdgcn_s_wait_tensorcnt)
#define HAVE_TDM 1
#else
#define HAVE_TDM 0
#endif

// ---------------- workspace layout (bytes) ----------------
static const size_t NP        = (size_t)NPIX;
static const size_t OFF_PATCH = 0;                                   // f32 [B][48][N]
static const size_t OFF_Y     = OFF_PATCH + 2ULL * 48 * NP * 4;      // bf16 [B][N][64]   (y / y2)
static const size_t OFF_QKV   = OFF_Y     + 2ULL * NP * 64 * 2;      // bf16 [B][144][N]  (later: h [B][256][N])
static const size_t OFF_QD    = OFF_QKV   + 2ULL * 144 * NP * 2;     // bf16 [B][96][N]   (q,k after dw)
static const size_t OFF_VCL   = OFF_QD    + 2ULL * 96 * NP * 2;      // bf16 [B][N][64]   (v channel-last)
static const size_t OFF_SPART = OFF_VCL   + 2ULL * NP * 64 * 2;      // f32 [B][512][2304]
static const size_t OFF_SFULL = OFF_SPART + 2ULL * 512 * 2304 * 4;   // f32 [B][2304]
static const size_t OFF_SSQ   = OFF_SFULL + 2ULL * 2304 * 4;         // f32 [B][96]
static const size_t OFF_MW    = OFF_SSQ   + 768;                     // bf16 [B][48][64]  (po@attn)
static const size_t OFF_WQ    = OFF_MW    + 2ULL * 48 * 64 * 2;      // bf16 [144][64]
static const size_t OFF_WPI   = OFF_WQ    + 144ULL * 64 * 2;         // bf16 [256][64]
static const size_t OFF_WFO   = OFF_WPI   + 256ULL * 64 * 2;         // bf16 [48][128]
// h (GDFN hidden, bf16 [B][256][N]) reuses OFF_QKV; g (bf16 [B][N][128]) reuses OFF_PATCH.

// ---------------- WMMA fragment helpers ----------------
union FragU { uint4 u[2]; v16bf v; };

// A operand, 16x32 (MxK), source row-major [M][ldk] bf16 (global or LDS).
// lane<16 : row m0+lane,    K = {k0..k0+7, k0+16..k0+23}
// lane>=16: row m0+lane-16, K = {k0+8..k0+15, k0+24..k0+31}
__device__ __forceinline__ v16bf load_afrag(const bf16_t* base, size_t ldk,
                                            int m0, int k0, int lane) {
  int row = m0 + (lane & 15);
  int kb  = k0 + ((lane >> 4) << 3);
  const bf16_t* p = base + (size_t)row * ldk + kb;
  FragU f;
  f.u[0] = *reinterpret_cast<const uint4*>(p);
  f.u[1] = *reinterpret_cast<const uint4*>(p + 16);
  return f.v;
}

// B operand, 32x16 (KxN), source: per output-column contiguous K ([col][ldk]).
// lane<16 : col c0+lane,    K = {k0..k0+7, k0+16..k0+23} ; lane>=16: col c0+lane-16, K = {k0+8..k0+15, k0+24..k0+31}
__device__ __forceinline__ v16bf load_bfrag(const bf16_t* base, size_t ldk,
                                            int c0, int k0, int lane) {
  int col = c0 + (lane & 15);
  int kb  = k0 + ((lane >> 4) << 3);
  const bf16_t* p = base + (size_t)col * ldk + kb;
  FragU f;
  f.u[0] = *reinterpret_cast<const uint4*>(p);
  f.u[1] = *reinterpret_cast<const uint4*>(p + 16);
  return f.v;
}

__device__ __forceinline__ v8f wmma_bf16(v16bf a, v16bf b, v8f c) {
  return __builtin_amdgcn_wmma_f32_16x16x32_bf16(false, a, false, b,
                                                 (short)0, c, false, false);
}

#if HAVE_TDM
// TDM: DMA a contiguous run of `nelem` bf16 (1-D tile, data_size=2B) from
// global memory into LDS.  D# built per CDNA5 ISA 8.3/8.4:
//   group0: count=1 | lds_addr | global_addr | type=2
//   group1: data_size=1(2B); tensor_dim0=tile_dim0=nelem; tensor_dim1=1;
//           tile_dim1=1; tensor_dim0_stride=nelem
__device__ __forceinline__ void tdm_load_lds(const bf16_t* gptr, bf16_t* lptr,
                                             int nelem) {
  unsigned lds_addr = (unsigned)(size_t)lptr;  // LDS aperture: addr[31:0] = LDS offset
  unsigned long long ga = (unsigned long long)(size_t)gptr;
  u32x4 g0 = { 1u,                                   // count=1 (valid user D#)
               lds_addr,
               (unsigned)ga,
               (unsigned)((ga >> 32) & 0x01FFFFFFu) | (2u << 30) };  // type=2
  i32x8 g1 = { 0x00010000,                           // data_size=1 -> 2 bytes
               (int)((nelem & 0xFFFF) << 16),        // tensor_dim0 lo
               (int)(((unsigned)nelem >> 16) | (1u << 16)),  // dim0 hi | tensor_dim1=1
               (int)(nelem << 16),                   // tile_dim0 = nelem (<=16384)
               1,                                    // tile_dim1=1, tile_dim2=0
               nelem,                                // tensor_dim0_stride lo
               0, 0 };
  i32x4 g2 = { 1, 1, 0, 0 };                         // tensor_dim2=1, tensor_dim3=1
  i32x4 g3 = { 0, 0, 0, 0 };
#if __clang_major__ >= 23
  i32x8 g4 = { 0, 0, 0, 0, 0, 0, 0, 0 };
  __builtin_amdgcn_tensor_load_to_lds(g0, g1, g2, g3, g4, 0);
#else
  __builtin_amdgcn_tensor_load_to_lds(g0, g1, g2, g3, 0);
#endif
}
#endif

// ---------------- weight prepack: f32 [M][K] -> bf16 [Mp][Kp] zero-padded ----
__global__ void prepack_k(const float* __restrict__ w, bf16_t* __restrict__ out,
                          int M, int K, int Mp, int Kp) {
  int i = blockIdx.x * 256 + threadIdx.x;
  if (i >= Mp * Kp) return;
  int mp = i / Kp, kp = i - mp * Kp;
  out[i] = (mp < M && kp < K) ? (bf16_t)w[mp * K + kp] : (bf16_t)0.0f;
}

// ---------------- 3x3 conv, 3->48 (patch embed), planar f32 out ------------
__global__ void __launch_bounds__(256)
patch_conv_k(const float* __restrict__ x, const float* __restrict__ pe_w,
             float* __restrict__ patch) {
  __shared__ float ws[48 * 27];
  for (int i = threadIdx.x; i < 48 * 27; i += 256) ws[i] = pe_w[i];
  __syncthreads();
  int n = blockIdx.x * 256 + threadIdx.x;
  int b = blockIdx.y;
  int h = n >> 9, w = n & 511;
  const float* xb = x + (size_t)b * 3 * NPIX;
  float in[27];
#pragma unroll
  for (int ci = 0; ci < 3; ++ci)
#pragma unroll
    for (int dy = 0; dy < 3; ++dy)
#pragma unroll
      for (int dx = 0; dx < 3; ++dx) {
        int hh = h + dy - 1, ww = w + dx - 1;
        bool ok = (hh >= 0) & (hh < HDIM) & (ww >= 0) & (ww < WDIM);
        in[ci * 9 + dy * 3 + dx] = ok ? xb[(size_t)ci * NPIX + hh * WDIM + ww] : 0.0f;
      }
  float* pb = patch + (size_t)b * 48 * NPIX + n;
  for (int co = 0; co < 48; ++co) {
    float a = 0.0f;
#pragma unroll
    for (int j = 0; j < 27; ++j) a += ws[co * 27 + j] * in[j];
    pb[(size_t)co * NPIX] = a;
  }
}

// ---------------- channel LayerNorm: planar f32 in -> channel-last bf16 [n][64]
__global__ void __launch_bounds__(256)
ln_k(const float* __restrict__ inp, const float* __restrict__ g,
     const float* __restrict__ be, bf16_t* __restrict__ out) {
  int n = blockIdx.x * 256 + threadIdx.x;
  int b = blockIdx.y;
  const float* p = inp + (size_t)b * 48 * NPIX + n;
  float v[48];
  float mu = 0.0f;
#pragma unroll
  for (int c = 0; c < 48; ++c) { v[c] = p[(size_t)c * NPIX]; mu += v[c]; }
  mu *= (1.0f / 48.0f);
  float var = 0.0f;
#pragma unroll
  for (int c = 0; c < 48; ++c) { float d = v[c] - mu; var += d * d; }
  var *= (1.0f / 48.0f);
  float rs = rsqrtf(var + 1e-5f);
  bf16_t* o = out + ((size_t)b * NPIX + n) * 64;
#pragma unroll
  for (int c = 0; c < 48; ++c) o[c] = (bf16_t)((v[c] - mu) * rs * g[c] + be[c]);
#pragma unroll
  for (int c = 48; c < 64; ++c) o[c] = (bf16_t)0.0f;
}

// ---------------- generic WMMA GEMM:  D[M, NPIX] = W[M,Kp] x X[NPIX,Kp]^T ---
// Weight panel staged into LDS via TDM (one DMA per workgroup).
// EPI 0: store bf16 planar   EPI 1: OutF = Res + acc (f32)   EPI 2: OutF += acc
template <int MTILES, int KCHUNKS, int EPI>
__global__ void __launch_bounds__(256)
gemm_k(const bf16_t* __restrict__ W, int wPerBatch,
       const bf16_t* __restrict__ X, bf16_t* __restrict__ OutB,
       float* __restrict__ OutF, const float* __restrict__ Res) {
  constexpr int KP   = KCHUNKS * 32;
  constexpr int WTOT = MTILES * 16 * KP;     // <= 16384 elements (32 KB)
  __shared__ alignas(16) bf16_t wlds[WTOT];

  const int b = blockIdx.y;
  const bf16_t* Wb = W + (size_t)b * wPerBatch;
#if HAVE_TDM
  if (threadIdx.x < 32) tdm_load_lds(Wb, wlds, WTOT);   // wave 0 issues the DMA
  __builtin_amdgcn_s_wait_tensorcnt((short)0);
  __syncthreads();
#else
  for (int i = threadIdx.x; i < WTOT; i += 256) wlds[i] = Wb[i];
  __syncthreads();
#endif

  const int lane = threadIdx.x & 31;
  const int wave = threadIdx.x >> 5;
  const int tile = blockIdx.x * 8 + wave;
  const int n0 = tile * 16;
  if (n0 >= NPIX) return;
  const bf16_t* Xb = X + (size_t)b * NPIX * KP;

  if (n0 + 128 < NPIX)
    __builtin_prefetch(Xb + ((size_t)n0 + 128) * KP, 0, 1);  // global_prefetch_b8

  v8f acc[MTILES] = {};
#pragma unroll
  for (int kc = 0; kc < KCHUNKS; ++kc) {
    const int k0 = kc * 32;
    v16bf bfrag = load_bfrag(Xb, KP, n0, k0, lane);
#pragma unroll
    for (int mt = 0; mt < MTILES; ++mt) {
      v16bf afrag = load_afrag(wlds, KP, mt * 16, k0, lane);  // ds_load_b128
      acc[mt] = wmma_bf16(afrag, bfrag, acc[mt]);
    }
  }
  const int nn = n0 + (lane & 15);
  const int mb = (lane >> 4) << 3;
#pragma unroll
  for (int mt = 0; mt < MTILES; ++mt) {
#pragma unroll
    for (int r = 0; r < 8; ++r) {
      int m = mt * 16 + mb + r;
      size_t off = ((size_t)b * MTILES * 16 + m) * NPIX + nn;
      if (EPI == 0)      OutB[off] = (bf16_t)acc[mt][r];
      else if (EPI == 1) OutF[off] = Res[off] + acc[mt][r];
      else               OutF[off] = OutF[off] + acc[mt][r];
    }
  }
}

// ---------------- depthwise 3x3 tap helper (planar bf16 input) -------------
__device__ __forceinline__ float dwtap(const bf16_t* ip, const float* wc,
                                       bool hm, bool hp, bool wm, bool wp) {
  float a = 0.0f;
  if (hm) {
    if (wm) a += wc[0] * (float)ip[-WDIM - 1];
    a += wc[1] * (float)ip[-WDIM];
    if (wp) a += wc[2] * (float)ip[-WDIM + 1];
  }
  if (wm) a += wc[3] * (float)ip[-1];
  a += wc[4] * (float)ip[0];
  if (wp) a += wc[5] * (float)ip[1];
  if (hp) {
    if (wm) a += wc[6] * (float)ip[WDIM - 1];
    a += wc[7] * (float)ip[WDIM];
    if (wp) a += wc[8] * (float)ip[WDIM + 1];
  }
  return a;
}

// dw 3x3 over qkv(144): q,k -> planar [B][96][N]; v -> channel-last [B][N][64]
__global__ void __launch_bounds__(256)
dw_qkv_k(const bf16_t* __restrict__ qkv, const float* __restrict__ dww,
         bf16_t* __restrict__ qd, bf16_t* __restrict__ vcl) {
  __shared__ float ws[144 * 9];
  for (int i = threadIdx.x; i < 144 * 9; i += 256) ws[i] = dww[i];
  __syncthreads();
  int n = blockIdx.x * 256 + threadIdx.x;
  int b = blockIdx.y;
  int h = n >> 9, w = n & 511;
  bool hm = h > 0, hp = h < HDIM - 1, wm = w > 0, wp = w < WDIM - 1;
  const bf16_t* inb = qkv + (size_t)b * 144 * NPIX + n;
  bf16_t* qb = qd + (size_t)b * 96 * NPIX + n;
  bf16_t* vb = vcl + ((size_t)b * NPIX + n) * 64;
  for (int c = 0; c < 144; ++c) {
    float a = dwtap(inb + (size_t)c * NPIX, ws + c * 9, hm, hp, wm, wp);
    if (c < 96) qb[(size_t)c * NPIX] = (bf16_t)a;
    else        vb[c - 96] = (bf16_t)a;
  }
#pragma unroll
  for (int j = 48; j < 64; ++j) vb[j] = (bf16_t)0.0f;
}

// per-channel sum of squares for q,k (deterministic single-block reduction)
__global__ void __launch_bounds__(256)
ssq_k(const bf16_t* __restrict__ qd, float* __restrict__ ssq) {
  int c = blockIdx.x, b = blockIdx.y;
  const bf16_t* p = qd + ((size_t)b * 96 + c) * NPIX;
  float s = 0.0f;
  for (int i = threadIdx.x; i < NPIX; i += 256) {
    float v = (float)p[i];
    s += v * v;
  }
  __shared__ float red[256];
  red[threadIdx.x] = s;
  __syncthreads();
  for (int st = 128; st > 0; st >>= 1) {
    if (threadIdx.x < st) red[threadIdx.x] += red[threadIdx.x + st];
    __syncthreads();
  }
  if (threadIdx.x == 0) ssq[b * 96 + c] = red[0];
}

// S = q . k^T over n (K=262144), WMMA, 512 waves/batch write f32 partials
#define SBLK  64
#define STOTW (SBLK * 8)              // 512 waves per batch
#define SITER (NPIX / 32 / STOTW)     // 16 k-chunks per wave

__global__ void __launch_bounds__(256)
sgemm_qk_k(const bf16_t* __restrict__ qd, float* __restrict__ Spart) {
  const int lane = threadIdx.x & 31;
  const int wave = threadIdx.x >> 5;
  const int b = blockIdx.y;
  const int gw = blockIdx.x * 8 + wave;
  const bf16_t* q = qd + (size_t)b * 96 * NPIX;
  const bf16_t* k = q + (size_t)48 * NPIX;
  v8f acc[9] = {};
  for (int it = 0; it < SITER; ++it) {
    int n0 = (gw * SITER + it) * 32;
    __builtin_prefetch(q + (size_t)n0 + 256, 0, 1);
    v16bf bf0 = load_bfrag(k, NPIX, 0, n0, lane);
    v16bf bf1 = load_bfrag(k, NPIX, 16, n0, lane);
    v16bf bf2 = load_bfrag(k, NPIX, 32, n0, lane);
#pragma unroll
    for (int ct = 0; ct < 3; ++ct) {
      v16bf af = load_afrag(q, NPIX, ct * 16, n0, lane);
      acc[ct * 3 + 0] = wmma_bf16(af, bf0, acc[ct * 3 + 0]);
      acc[ct * 3 + 1] = wmma_bf16(af, bf1, acc[ct * 3 + 1]);
      acc[ct * 3 + 2] = wmma_bf16(af, bf2, acc[ct * 3 + 2]);
    }
  }
  float* outp = Spart + ((size_t)b * STOTW + gw) * 2304;
  const int mb = (lane >> 4) << 3;
#pragma unroll
  for (int ct = 0; ct < 3; ++ct)
#pragma unroll
    for (int dt = 0; dt < 3; ++dt)
#pragma unroll
      for (int r = 0; r < 8; ++r)
        outp[(ct * 16 + mb + r) * 48 + dt * 16 + (lane & 15)] = acc[ct * 3 + dt][r];
}

__global__ void s_reduce_k(const float* __restrict__ Spart, float* __restrict__ Sfull) {
  int i = blockIdx.x * 256 + threadIdx.x;
  int b = blockIdx.y;
  if (i >= 2304) return;
  float s = 0.0f;
  for (int g = 0; g < STOTW; ++g) s += Spart[((size_t)b * STOTW + g) * 2304 + i];
  Sfull[b * 2304 + i] = s;
}

// scale by L2 norms + temp, softmax rows, then fold M = po_w @ attn (bf16 padded)
__global__ void __launch_bounds__(64)
softmax_m_k(const float* __restrict__ Sfull, const float* __restrict__ ssq,
            const float* __restrict__ temp, const float* __restrict__ po_w,
            bf16_t* __restrict__ Mw) {
  int b = blockIdx.x, t = threadIdx.x;
  __shared__ float attn[48 * 48];
  if (t < 48) {
    float nq = fmaxf(sqrtf(ssq[b * 96 + t]), 1e-12f);
    float tp = temp[0];
    float row[48];
    float mx = -1e30f;
    for (int d = 0; d < 48; ++d) {
      float nk = fmaxf(sqrtf(ssq[b * 96 + 48 + d]), 1e-12f);
      float v = Sfull[b * 2304 + t * 48 + d] / (nq * nk) * tp;
      row[d] = v;
      mx = fmaxf(mx, v);
    }
    float s = 0.0f;
    for (int d = 0; d < 48; ++d) { float e = expf(row[d] - mx); row[d] = e; s += e; }
    float inv = 1.0f / s;
    for (int d = 0; d < 48; ++d) attn[t * 48 + d] = row[d] * inv;
  }
  __syncthreads();
  if (t < 48) {
    for (int d = 0; d < 64; ++d) {
      float m = 0.0f;
      if (d < 48)
        for (int c = 0; c < 48; ++c) m += po_w[t * 48 + c] * attn[c * 48 + d];
      Mw[((size_t)b * 48 + t) * 64 + d] = (bf16_t)m;
    }
  }
}

// GDFN depthwise + exact-GELU gate: h planar [B][256][N] -> g channel-last [B][N][128]
__global__ void __launch_bounds__(256)
dw_gate_k(const bf16_t* __restrict__ hbuf, const float* __restrict__ dww,
          bf16_t* __restrict__ g) {
  __shared__ float ws[254 * 9];
  for (int i = threadIdx.x; i < 254 * 9; i += 256) ws[i] = dww[i];
  __syncthreads();
  int n = blockIdx.x * 256 + threadIdx.x;
  int b = blockIdx.y;
  int h = n >> 9, w = n & 511;
  bool hm = h > 0, hp = h < HDIM - 1, wm = w > 0, wp = w < WDIM - 1;
  const bf16_t* inb = hbuf + (size_t)b * 256 * NPIX + n;
  bf16_t* gb = g + ((size_t)b * NPIX + n) * 128;
  for (int c = 0; c < 127; ++c) {
    float a = dwtap(inb + (size_t)c * NPIX, ws + c * 9, hm, hp, wm, wp);
    float x2 = dwtap(inb + (size_t)(c + 127) * NPIX, ws + (c + 127) * 9, hm, hp, wm, wp);
    float ge = 0.5f * a * (1.0f + erff(a * 0.70710678118654752f));
    gb[c] = (bf16_t)(ge * x2);
  }
  gb[127] = (bf16_t)0.0f;
}

// ---------------------------------------------------------------------------
extern "C" void kernel_launch(void* const* d_in, const int* in_sizes, int n_in,
                              void* d_out, int out_size, void* d_ws, size_t ws_size,
                              hipStream_t stream) {
  (void)in_sizes; (void)n_in; (void)out_size; (void)ws_size;
  const float* x      = (const float*)d_in[0];
  const float* pe_w   = (const float*)d_in[1];
  const float* n1w    = (const float*)d_in[2];
  const float* n1b    = (const float*)d_in[3];
  const float* temp   = (const float*)d_in[4];
  const float* qkv_w  = (const float*)d_in[5];
  const float* qkv_dw = (const float*)d_in[6];
  const float* po_w   = (const float*)d_in[7];
  const float* n2w    = (const float*)d_in[8];
  const float* n2b    = (const float*)d_in[9];
  const float* pi_w   = (const float*)d_in[10];
  const float* ffn_dw = (const float*)d_in[11];
  const float* fo_w   = (const float*)d_in[12];
  float* out = (float*)d_out;

  char* ws = (char*)d_ws;
  float*  patch = (float*)(ws + OFF_PATCH);
  bf16_t* ybuf  = (bf16_t*)(ws + OFF_Y);
  bf16_t* qkvb  = (bf16_t*)(ws + OFF_QKV);
  bf16_t* qdb   = (bf16_t*)(ws + OFF_QD);
  bf16_t* vclb  = (bf16_t*)(ws + OFF_VCL);
  bf16_t* hbuf  = (bf16_t*)(ws + OFF_QKV);   // reuse: qkv/qd/vcl dead by then
  bf16_t* gbuf  = (bf16_t*)(ws + OFF_PATCH); // reuse: patch/y dead by then
  float*  spart = (float*)(ws + OFF_SPART);
  float*  sfull = (float*)(ws + OFF_SFULL);
  float*  ssqb  = (float*)(ws + OFF_SSQ);
  bf16_t* mw    = (bf16_t*)(ws + OFF_MW);
  bf16_t* wq    = (bf16_t*)(ws + OFF_WQ);
  bf16_t* wpi   = (bf16_t*)(ws + OFF_WPI);
  bf16_t* wfo   = (bf16_t*)(ws + OFF_WFO);

  dim3 blk(256);
  dim3 gpix(NPIX / 256, NB);
  dim3 ggemm(NPIX / 16 / 8, NB);

  // weight prepack to bf16, zero-padded
  prepack_k<<<(144 * 64 + 255) / 256, blk, 0, stream>>>(qkv_w, wq, 144, 48, 144, 64);
  prepack_k<<<(256 * 64 + 255) / 256, blk, 0, stream>>>(pi_w, wpi, 254, 48, 256, 64);
  prepack_k<<<(48 * 128 + 255) / 256, blk, 0, stream>>>(fo_w, wfo, 48, 127, 48, 128);

  // patch embed + LN1
  patch_conv_k<<<gpix, blk, 0, stream>>>(x, pe_w, patch);
  ln_k<<<gpix, blk, 0, stream>>>(patch, n1w, n1b, ybuf);

  // MDTA
  gemm_k<9, 2, 0><<<ggemm, blk, 0, stream>>>(wq, 0, ybuf, qkvb, nullptr, nullptr);
  dw_qkv_k<<<gpix, blk, 0, stream>>>(qkvb, qkv_dw, qdb, vclb);
  ssq_k<<<dim3(96, NB), blk, 0, stream>>>(qdb, ssqb);
  sgemm_qk_k<<<dim3(SBLK, NB), blk, 0, stream>>>(qdb, spart);
  s_reduce_k<<<dim3(9, NB), blk, 0, stream>>>(spart, sfull);
  softmax_m_k<<<NB, 64, 0, stream>>>(sfull, ssqb, temp, po_w, mw);
  // first = patch + (po@attn) @ v   -> d_out (planar f32)
  gemm_k<3, 2, 1><<<ggemm, blk, 0, stream>>>(mw, 48 * 64, vclb, nullptr, out, patch);

  // GDFN
  ln_k<<<gpix, blk, 0, stream>>>(out, n2w, n2b, ybuf);
  gemm_k<16, 2, 0><<<ggemm, blk, 0, stream>>>(wpi, 0, ybuf, hbuf, nullptr, nullptr);
  dw_gate_k<<<gpix, blk, 0, stream>>>(hbuf, ffn_dw, gbuf);
  // out += fo_w @ g
  gemm_k<3, 4, 2><<<ggemm, blk, 0, stream>>>(wfo, 0, gbuf, nullptr, out, nullptr);
}
